// SubModel_22016002359901
// MI455X (gfx1250) — matrix operand order repaired and checked by reading, not compiled, not run on hardware
//
#include <hip/hip_runtime.h>
#include <math.h>

// ---------------------------------------------------------------------------
// CDNA5 (gfx1250) implementation of the 3-branch GCN + MLP stack.
//   - All dense GEMMs use v_wmma_f32_16x16x32_f16 (f16 inputs, f32 accum).
//   - Each wave computes a 16x64 output tile (4 WMMA accumulators) so the A
//     fragment is loaded/converted once per K-step and reused 4x.
//   - Weights are pre-converted to f16, transposed so each lane's B fragment
//     is two contiguous 16B loads (matches ISA 7.12.2 B-matrix layout).
//   - Edge aggregation: wave-per-edge float4 atomic scatter (L2-resident H).
// ---------------------------------------------------------------------------

typedef __attribute__((ext_vector_type(16))) _Float16 v16h;
typedef __attribute__((ext_vector_type(8)))  float    v8f;

// ---- weight convert + transpose: Wt[m][k] = (half) W[k][m] ---------------
__global__ void cvt_w_kernel(const float* __restrict__ W, _Float16* __restrict__ Wt,
                             int K, int M) {
    int idx = blockIdx.x * blockDim.x + threadIdx.x;
    if (idx >= K * M) return;
    int k = idx / M;
    int m = idx - k * M;
    Wt[(size_t)m * K + k] = (_Float16)W[idx];
}

// ---- WMMA GEMM: C[N,M] = relu?( A[N,K] @ W[K,M] + bias ), opt C2 += C -----
// One wave computes one 16x64 tile (4 accumulators). N rows must be a
// multiple of 16 (100000 = 6250*16), K a multiple of 32, M a multiple of 64.
template <bool RELU, bool ADD2>
__global__ void gemm_wmma_kernel(const float* __restrict__ A,
                                 const _Float16* __restrict__ Wt,   // [M][K] f16
                                 const float* __restrict__ bias,    // [M] or null
                                 float* __restrict__ C,
                                 float* __restrict__ C2,
                                 int Nrows, int K, int M) {
    const int lane      = threadIdx.x & 31;
    const int waveId    = blockIdx.x * (blockDim.x >> 5) + (threadIdx.x >> 5);
    const int colGroups = M >> 6;                     // 64-wide column groups
    const int rowTile   = waveId / colGroups;
    const int colGrp    = waveId - rowTile * colGroups;
    if (rowTile * 16 >= Nrows) return;    // wave-uniform exit (EXEC stays all-1)

    const int n16 = lane & 15;            // row-in-tile for A, col-in-tile for B/C
    const int kh  = lane >> 4;            // K-half selector per ISA A/B layouts
    const float*    __restrict__ arow = A + (size_t)(rowTile * 16 + n16) * K;
    const int       col0              = colGrp * 64 + n16;
    const _Float16* __restrict__ wcol = Wt + (size_t)col0 * K;   // +c*16*K per sub-tile

    v8f acc0 = {0.f,0.f,0.f,0.f,0.f,0.f,0.f,0.f};
    v8f acc1 = acc0, acc2 = acc0, acc3 = acc0;

    for (int kt = 0; kt < K; kt += 32) {
        // A fragment (16-bit A layout 16x32): lane kh=0 -> K 0-7 & 16-23,
        // kh=1 -> K 8-15 & 24-31. Two contiguous 8-float runs, cvt to f16.
        const float4 f0 = *(const float4*)(arow + kt + kh * 8);
        const float4 f1 = *(const float4*)(arow + kt + kh * 8 + 4);
        const float4 f2 = *(const float4*)(arow + kt + 16 + kh * 8);
        const float4 f3 = *(const float4*)(arow + kt + 16 + kh * 8 + 4);
        v16h a;
        a[0]  = (_Float16)f0.x; a[1]  = (_Float16)f0.y; a[2]  = (_Float16)f0.z; a[3]  = (_Float16)f0.w;
        a[4]  = (_Float16)f1.x; a[5]  = (_Float16)f1.y; a[6]  = (_Float16)f1.z; a[7]  = (_Float16)f1.w;
        a[8]  = (_Float16)f2.x; a[9]  = (_Float16)f2.y; a[10] = (_Float16)f2.z; a[11] = (_Float16)f2.w;
        a[12] = (_Float16)f3.x; a[13] = (_Float16)f3.y; a[14] = (_Float16)f3.z; a[15] = (_Float16)f3.w;

        // B fragments: same K order as A: kh=0 -> K 0-7 & 16-23,
        // kh=1 -> K 8-15 & 24-31 -> two 16B loads per sub-tile.
        typedef __attribute__((ext_vector_type(8))) _Float16 v8h_t;
#define LDB(c) __builtin_shufflevector(*(const v8h_t*)(wcol + (size_t)(c) * 16 * K + kt + kh * 8), *(const v8h_t*)(wcol + (size_t)(c) * 16 * K + kt + 16 + kh * 8), 0,1,2,3,4,5,6,7,8,9,10,11,12,13,14,15)
        const v16h b0 = LDB(0);
        const v16h b1 = LDB(1);
        const v16h b2 = LDB(2);
        const v16h b3 = LDB(3);
#undef LDB

        acc0 = __builtin_amdgcn_wmma_f32_16x16x32_f16(false, a, false, b0, (short)0, acc0, false, false);
        acc1 = __builtin_amdgcn_wmma_f32_16x16x32_f16(false, a, false, b1, (short)0, acc1, false, false);
        acc2 = __builtin_amdgcn_wmma_f32_16x16x32_f16(false, a, false, b2, (short)0, acc2, false, false);
        acc3 = __builtin_amdgcn_wmma_f32_16x16x32_f16(false, a, false, b3, (short)0, acc3, false, false);
    }

    // C/D layout: VGPR r -> row (kh*8 + r), col n16 (within each 16-col tile).
    const int mbase = kh * 8;
    v8f accs[4] = {acc0, acc1, acc2, acc3};
#pragma unroll
    for (int c = 0; c < 4; ++c) {
        const int   col = col0 + c * 16;
        const float bv  = bias ? bias[col] : 0.0f;
#pragma unroll
        for (int r = 0; r < 8; ++r) {
            float v = accs[c][r] + bv;
            if (RELU) v = fmaxf(v, 0.0f);
            const size_t idx = (size_t)(rowTile * 16 + mbase + r) * M + col;
            C[idx] = v;
            if (ADD2) C2[idx] += v;
        }
    }
}

// ---- degree accumulation -------------------------------------------------
__global__ void deg_kernel(const int* __restrict__ dst, const float* __restrict__ ew,
                           float* __restrict__ deg, int E) {
    int e = blockIdx.x * blockDim.x + threadIdx.x;
    if (e < E) atomicAdd(&deg[dst[e]], ew[e]);
}

__global__ void dinv_kernel(float* __restrict__ deg, int n) {
    int i = blockIdx.x * blockDim.x + threadIdx.x;
    if (i < n) deg[i] = rsqrtf(deg[i] + 1.0f);   // in-place: deg -> dinv
}

// ---- edge scatter: AGG[dst] += H[src] * (dinv[src]*ew*dinv[dst]) ---------
// One wave per edge; each lane handles 4 of the 128 features (float4).
__global__ void agg_kernel(const float* __restrict__ H,
                           const int* __restrict__ src, const int* __restrict__ dst,
                           const float* __restrict__ ew, const float* __restrict__ dinv,
                           float* __restrict__ AGG, int E) {
    const long long tid = (long long)blockIdx.x * blockDim.x + threadIdx.x;
    const int e = (int)(tid >> 5);
    if (e >= E) return;
    const int lane = threadIdx.x & 31;
    const int s = src[e];
    const int d = dst[e];
    const float norm = dinv[s] * ew[e] * dinv[d];
    const float4 hv = *(const float4*)(H + (size_t)s * 128 + lane * 4);
    float* out = AGG + (size_t)d * 128 + lane * 4;
    atomicAdd(out + 0, hv.x * norm);
    atomicAdd(out + 1, hv.y * norm);
    atomicAdd(out + 2, hv.z * norm);
    atomicAdd(out + 3, hv.w * norm);
}

// ---- self-loop term + bias + relu (in place on AGG) ----------------------
__global__ void finish_conv_kernel(float* __restrict__ AGG, const float* __restrict__ H,
                                   const float* __restrict__ dinv,
                                   const float* __restrict__ bias, int total) {
    int idx = blockIdx.x * blockDim.x + threadIdx.x;
    if (idx >= total) return;
    const int i = idx >> 7;      // node
    const int f = idx & 127;     // feature
    const float di = dinv[i];
    AGG[idx] = fmaxf(AGG[idx] + di * di * H[idx] + bias[f], 0.0f);
}

// ---------------------------------------------------------------------------
extern "C" void kernel_launch(void* const* d_in, const int* in_sizes, int n_in,
                              void* d_out, int out_size, void* d_ws, size_t ws_size,
                              hipStream_t stream) {
    (void)n_in; (void)out_size; (void)ws_size;
    const int   F = 128;
    const int   N = in_sizes[0] / F;           // 100000 (multiple of 16)
    const float* X = (const float*)d_in[0];

    // input indices (setup_inputs insertion order)
    // 0:X | per g in 0..2: 1+6g:edge_index 2+6g:edge_weight 3+6g:w1 4+6g:b1 5+6g:w2 6+6g:b2
    // 19..24: fc1 w1,b1,w2,b2,w3,b3 | 25..30: fc2 w1,b1,w2,b2,w3,b3
    const float* fc1_w[3] = {(const float*)d_in[19], (const float*)d_in[21], (const float*)d_in[23]};
    const float* fc1_b[3] = {(const float*)d_in[20], (const float*)d_in[22], (const float*)d_in[24]};
    const float* fc2_w[3] = {(const float*)d_in[25], (const float*)d_in[27], (const float*)d_in[29]};
    const float* fc2_b[3] = {(const float*)d_in[26], (const float*)d_in[28], (const float*)d_in[30]};

    // ---- workspace layout ----
    char* ws = (char*)d_ws;
    _Float16* hw = (_Float16*)ws;              // f16 transposed-weight region
    size_t off = 0;
    _Float16* g_w1t[3];
    _Float16* g_w2t[3];
    for (int g = 0; g < 3; ++g) {
        g_w1t[g] = hw + off; off += (size_t)F * F;
        g_w2t[g] = hw + off; off += (size_t)F * F;
    }
    _Float16* fc1_wt[3];
    _Float16* fc2_wt[3];
    const int fcK[3] = {128, 256, 128};
    const int fcM[3] = {256, 128, 64};
    for (int l = 0; l < 3; ++l) { fc1_wt[l] = hw + off; off += (size_t)fcK[l] * fcM[l]; }
    for (int l = 0; l < 3; ++l) { fc2_wt[l] = hw + off; off += (size_t)fcK[l] * fcM[l]; }

    float* B0   = (float*)(ws + (512 << 10));      // N x 128
    float* B1   = B0 + (size_t)N * F;              // N x 128 (B0,B1 contiguous -> N x 256)
    float* B2   = B1 + (size_t)N * F;              // N x 128
    float* dinv = B2 + (size_t)N * F;              // N
    float* T1   = B0;                              // N x 256 alias over B0+B1

    float* Xs   = (float*)d_out;                   // [N,64]
    float* Xout = (float*)d_out;                   // Xout + (1+g)*N*64 = X_{g+1}

    // ---- helpers ----
    auto cvt = [&](const float* W, _Float16* Wt, int K, int M) {
        int total = K * M;
        cvt_w_kernel<<<(total + 255) / 256, 256, 0, stream>>>(W, Wt, K, M);
    };
    auto gemm = [&](const float* A, const _Float16* Wt, const float* bias,
                    float* C, float* C2, int K, int M, bool relu, bool add2) {
        const int rowTiles = (N + 15) >> 4;
        const long long waves = (long long)rowTiles * (M >> 6);   // 16x64 per wave
        const int blocks = (int)((waves + 7) >> 3);               // 8 waves / block
        if (relu && add2)
            gemm_wmma_kernel<true,  true ><<<blocks, 256, 0, stream>>>(A, Wt, bias, C, C2, N, K, M);
        else if (relu)
            gemm_wmma_kernel<true,  false><<<blocks, 256, 0, stream>>>(A, Wt, bias, C, nullptr, N, K, M);
        else
            gemm_wmma_kernel<false, false><<<blocks, 256, 0, stream>>>(A, Wt, bias, C, nullptr, N, K, M);
    };

    // ---- 1) convert all weights to transposed f16 ----
    for (int g = 0; g < 3; ++g) {
        cvt((const float*)d_in[3 + 6 * g], g_w1t[g], F, F);
        cvt((const float*)d_in[5 + 6 * g], g_w2t[g], F, F);
    }
    for (int l = 0; l < 3; ++l) {
        cvt(fc1_w[l], fc1_wt[l], fcK[l], fcM[l]);
        cvt(fc2_w[l], fc2_wt[l], fcK[l], fcM[l]);
    }

    // ---- 2) fc1 branch: Xs = fc_layer(X) (initializes Xs) ----
    gemm(X,  fc1_wt[0], fc1_b[0], T1, nullptr, 128, 256, true, false);
    gemm(T1, fc1_wt[1], fc1_b[1], B2, nullptr, 256, 128, true, false);
    gemm(B2, fc1_wt[2], fc1_b[2], Xs, nullptr, 128, 64,  true, false);

    // ---- 3) three GCN modules ----
    for (int g = 0; g < 3; ++g) {
        const int*   ei  = (const int*)  d_in[1 + 6 * g];
        const float* ew  = (const float*)d_in[2 + 6 * g];
        const float* b1  = (const float*)d_in[4 + 6 * g];
        const float* b2  = (const float*)d_in[6 + 6 * g];
        const int    E   = in_sizes[2 + 6 * g];
        const int*   src = ei;
        const int*   dst = ei + E;
        const int degBlocks = (E + 255) / 256;
        const int aggBlocks = (int)(((long long)E * 32 + 255) / 256);
        const int nfBlocks  = (N * F + 255) / 256;

        // degree -> dinv (shared by both convs of this module)
        hipMemsetAsync(dinv, 0, (size_t)N * sizeof(float), stream);
        deg_kernel <<<degBlocks, 256, 0, stream>>>(dst, ew, dinv, E);
        dinv_kernel<<<(N + 255) / 256, 256, 0, stream>>>(dinv, N);

        // conv1: H = X @ w1 ; AGG = scatter ; += self-loop + bias ; relu
        gemm(X, g_w1t[g], nullptr, B0, nullptr, F, F, false, false);
        hipMemsetAsync(B1, 0, (size_t)N * F * sizeof(float), stream);
        agg_kernel<<<aggBlocks, 256, 0, stream>>>(B0, src, dst, ew, dinv, B1, E);
        finish_conv_kernel<<<nfBlocks, 256, 0, stream>>>(B1, B0, dinv, b1, N * F);

        // conv2: H = X1 @ w2 ; AGG ; epilogue
        gemm(B1, g_w2t[g], nullptr, B0, nullptr, F, F, false, false);
        hipMemsetAsync(B2, 0, (size_t)N * F * sizeof(float), stream);
        agg_kernel<<<aggBlocks, 256, 0, stream>>>(B0, src, dst, ew, dinv, B2, E);
        finish_conv_kernel<<<nfBlocks, 256, 0, stream>>>(B2, B0, dinv, b2, N * F);

        // fc2 branch: X_g -> d_out slot, and Xs += X_g
        gemm(B2, fc2_wt[0], fc2_b[0], T1, nullptr, 128, 256, true, false);
        gemm(T1, fc2_wt[1], fc2_b[1], B2, nullptr, 256, 128, true, false);
        gemm(B2, fc2_wt[2], fc2_b[2], Xout + (size_t)(1 + g) * N * 64, Xs,
             128, 64, true, true);
    }
}
